// Net_SLSTM_88553635709490
// MI455X (gfx1250) — matrix operand order repaired and checked by reading, not compiled, not run on hardware
//
#include <hip/hip_runtime.h>

// ---------------------------------------------------------------------------
// Types for CDNA5 WMMA bf16 (v_wmma_f32_16x16x32_bf16)
// ---------------------------------------------------------------------------
typedef __attribute__((ext_vector_type(16))) __bf16 v16bf;
typedef __attribute__((ext_vector_type(8)))  float  v8f;

union BF16x16 { v16bf v; unsigned short u[16]; };

__device__ __forceinline__ unsigned short f2bf(float f) {
  unsigned int u = __float_as_uint(f);
  u += 0x7FFFu + ((u >> 16) & 1u);      // round-to-nearest-even
  return (unsigned short)(u >> 16);
}

// Fast activations: v_rcp_f32 + v_exp_f32 instead of IEEE divide / libm tanh.
// Saturation behavior is correct (exp->inf => rcp->0).
__device__ __forceinline__ float sigm(float x) {
  return __builtin_amdgcn_rcpf(1.0f + __expf(-x));
}
__device__ __forceinline__ float fast_tanh(float x) {
  return 1.0f - 2.0f * __builtin_amdgcn_rcpf(1.0f + __expf(2.0f * x));
}

// ---------------------------------------------------------------------------
// K1: per-channel sum / sumsq of x [T*B, 14] for input BatchNorm
// stats layout: [0..13] sum, [16..29] sumsq
// ---------------------------------------------------------------------------
__global__ __launch_bounds__(256) void bn_stats_kernel(
    const float* __restrict__ x, float* __restrict__ stats) {
  __shared__ float red[256];
  const int tid = threadIdx.x;
  float s[14], sq[14];
#pragma unroll
  for (int c = 0; c < 14; ++c) { s[c] = 0.f; sq[c] = 0.f; }
  const size_t N = 400ull * 1024ull;
  for (size_t n = (size_t)blockIdx.x * blockDim.x + tid; n < N;
       n += (size_t)gridDim.x * blockDim.x) {
    const float* row = x + n * 14;
#pragma unroll
    for (int c = 0; c < 14; ++c) { float v = row[c]; s[c] += v; sq[c] += v * v; }
  }
  for (int c = 0; c < 14; ++c) {
    red[tid] = s[c]; __syncthreads();
    for (int off = 128; off; off >>= 1) {
      if (tid < off) red[tid] += red[tid + off];
      __syncthreads();
    }
    if (tid == 0) atomicAdd(&stats[c], red[0]);
    __syncthreads();
    red[tid] = sq[c]; __syncthreads();
    for (int off = 128; off; off >>= 1) {
      if (tid < off) red[tid] += red[tid + off];
      __syncthreads();
    }
    if (tid == 0) atomicAdd(&stats[16 + c], red[0]);
    __syncthreads();
  }
}

// ---------------------------------------------------------------------------
// K2: fold input BN into layer-1 input weights.
//   a1 = g*rsqrt(var+eps); c1 = b - mu*a1
//   W1s[k][j] (bf16, k<32 padded) = a1[k]*Wih1[j,k]
//   cb1[j] = sum_c c1[c]*Wih1[j,c] + bih1[j] + bhh1[j]
// ---------------------------------------------------------------------------
__global__ __launch_bounds__(256) void prep1_kernel(
    const float* __restrict__ stats, const float* __restrict__ Wih1,
    const float* __restrict__ bih1, const float* __restrict__ bhh1,
    const float* __restrict__ g, const float* __restrict__ b,
    float* __restrict__ cb1, unsigned short* __restrict__ W1s) {
  __shared__ float a1[14], c1[14];
  const int tid = threadIdx.x;
  const float invN = 1.0f / 409600.0f;
  if (tid < 14) {
    float mu  = stats[tid] * invN;
    float var = stats[16 + tid] * invN - mu * mu;
    float a   = g[tid] * rsqrtf(var + 1e-5f);
    a1[tid] = a;
    c1[tid] = b[tid] - mu * a;
  }
  __syncthreads();
  for (int j = tid; j < 512; j += 256) {
    float acc = bih1[j] + bhh1[j];
#pragma unroll
    for (int c = 0; c < 14; ++c) acc += c1[c] * Wih1[j * 14 + c];
    cb1[j] = acc;
  }
  for (int idx = tid; idx < 32 * 512; idx += 256) {
    int k = idx >> 9, j = idx & 511;
    float v = (k < 14) ? a1[k] * Wih1[j * 14 + k] : 0.f;
    W1s[idx] = f2bf(v);
  }
}

// ---------------------------------------------------------------------------
// K4: fold spike BN into layer-2 input weights. Spikes are 0/1 so
//   p = sum/N ; var = p - p^2
// ---------------------------------------------------------------------------
__global__ __launch_bounds__(256) void prep2_kernel(
    const float* __restrict__ spk_sum, const float* __restrict__ Wih2,
    const float* __restrict__ bih2, const float* __restrict__ bhh2,
    const float* __restrict__ g, const float* __restrict__ b,
    float* __restrict__ cb2, unsigned short* __restrict__ W2s) {
  __shared__ float a2[128], c2[128];
  const int tid = threadIdx.x;
  if (tid < 128) {
    float p   = spk_sum[tid] * (1.0f / 409600.0f);
    float var = p - p * p;
    float a   = g[tid] * rsqrtf(var + 1e-5f);
    a2[tid] = a;
    c2[tid] = b[tid] - p * a;
  }
  __syncthreads();
  for (int j = tid; j < 512; j += 256) {
    float acc = bih2[j] + bhh2[j];
#pragma unroll 8
    for (int k = 0; k < 128; ++k) acc += c2[k] * Wih2[j * 128 + k];
    cb2[j] = acc;
  }
  for (int idx = tid; idx < 128 * 512; idx += 256) {
    int k = idx >> 9, col = idx & 511;
    W2s[idx] = f2bf(a2[k] * Wih2[col * 128 + k]);
  }
}

// ---------------------------------------------------------------------------
// K3 / K5: persistent SLSTM scan. One workgroup owns 16 batch rows for the
// full T=400 scan. 8 waves each own a 64-column slice of the 512 gates
// (so each wave handles exactly one gate nonlinearity). Recurrent weights
// (and layer-2 input weights) are held stationary in VGPRs as WMMA B
// fragments; the 16x128 mem tile cycles through LDS each step.
// ---------------------------------------------------------------------------
template <int LAYER>
__global__ __launch_bounds__(256) void slstm_layer_kernel(
    const float* __restrict__ xin,              // L1: x [T,B,14]
    const unsigned short* __restrict__ spk_in,  // L2: spk1 bf16 [T,B,128]
    const float* __restrict__ Whh,              // [512,128]
    const unsigned short* __restrict__ WinS,    // bf16 [KIN][512]
    const float* __restrict__ cb,               // [512] fused bias
    const float* __restrict__ thr_p,
    unsigned short* __restrict__ spk_out,       // L1: bf16 [T,B,128]
    float* __restrict__ spk_sum,                // L1: [128]
    float* __restrict__ mem_out)                // L2: [B,128] mean mem
{
  constexpr int T = 400, B = 1024, H = 128, G = 512;
  constexpr int NKIN = (LAYER == 1) ? 1 : 4;    // input K chunks of 32

  const int tid  = threadIdx.x;
  const int wave = tid >> 5;
  const int lane = tid & 31;
  const int r0   = blockIdx.x * 16;             // batch-row tile base
  const float thr = thr_p[0];

  __shared__ unsigned short s_memA[16 * 128];   // prev mem, bf16 (A matrix)
  __shared__ unsigned short s_inA[16 * 128];    // input tile, bf16 (A matrix)
  __shared__ float s_act[16 * 512];             // activated gates

  // Fragment index helpers (ISA 7.12.2 layouts, wave32)
  const int nl  = lane & 15;       // N / A-row index
  const int hi  = lane >> 4;       // lane half
  const int kbB = hi << 4;         // unused K base (0/16); B frags use kbA
  const int kbA = hi << 3;         // A-frag K base (0/8)

  // ---- Stationary B fragments in VGPRs -----------------------------------
  BF16x16 whhF[4][4];              // [n-tile][k-chunk]
#pragma unroll
  for (int nt = 0; nt < 4; ++nt) {
    const int col = wave * 64 + nt * 16 + nl;
#pragma unroll
    for (int kc = 0; kc < 4; ++kc)
#pragma unroll
      for (int j = 0; j < 16; ++j)
        whhF[nt][kc].u[j] = f2bf(Whh[col * H + kc * 32 + kbA + j + ((j >= 8) ? 8 : 0)]);
  }
  BF16x16 winF[4][NKIN];
#pragma unroll
  for (int nt = 0; nt < 4; ++nt) {
    const int col = wave * 64 + nt * 16 + nl;
#pragma unroll
    for (int kc = 0; kc < NKIN; ++kc)
#pragma unroll
      for (int j = 0; j < 16; ++j)
        winF[nt][kc].u[j] = WinS[(kc * 32 + kbA + j + ((j >= 8) ? 8 : 0)) * G + col];
  }
  v8f biasF[4];
#pragma unroll
  for (int nt = 0; nt < 4; ++nt) {
    const float bv = cb[wave * 64 + nt * 16 + nl];
#pragma unroll
    for (int j = 0; j < 8; ++j) biasF[nt][j] = bv;
  }

  // ---- Thread-owned recurrent state (8 contiguous h per thread) ----------
  const int e0   = tid * 8;
  const int srow = e0 >> 7;        // 0..15
  const int sh   = e0 & 127;       // h base
  float syn[8], mem[8], accum[8];
#pragma unroll
  for (int j = 0; j < 8; ++j) { syn[j] = 0.f; mem[j] = 0.f; accum[j] = 0.f; }
#pragma unroll
  for (int j = 0; j < 8; ++j) s_memA[srow * H + sh + j] = 0;
  __syncthreads();

  const int gsel = wave >> 1;      // 0:i 1:f 2:g(tanh) 3:o

  for (int t = 0; t < T; ++t) {
    // -- stage input A tile into LDS --
    if constexpr (LAYER == 1) {
      int slot = tid * 2;
#pragma unroll
      for (int s = 0; s < 2; ++s, ++slot) {
        const int row = slot >> 5, c = slot & 31;
        const float xv =
            (c < 14) ? xin[((size_t)t * B + r0 + row) * 14 + c] : 0.f;
        s_inA[row * 32 + c] = f2bf(xv);
      }
    } else {
      const int row = tid >> 4, seg = tid & 15;
      const uint4* src =
          (const uint4*)(spk_in + ((size_t)t * B + r0 + row) * H);
      ((uint4*)(s_inA + row * H))[seg] = src[seg];
      if (t + 1 < T)
        __builtin_prefetch(
            spk_in + ((size_t)(t + 1) * B + r0 + row) * H + seg * 8, 0, 0);
    }
    __syncthreads();

    // -- gates = bias + input@WinS + mem@Whh  (WMMA, bf16 -> f32) --
    v8f acc[4];
    {
      // first input chunk consumes the bias fragment directly as C
      BF16x16 aF;
#pragma unroll
      for (int j = 0; j < 16; ++j) {
        const int K = kbA + j + ((j >= 8) ? 8 : 0);
        aF.u[j] = (LAYER == 1) ? s_inA[nl * 32 + K] : s_inA[nl * H + K];
      }
#pragma unroll
      for (int nt = 0; nt < 4; ++nt)
        acc[nt] = __builtin_amdgcn_wmma_f32_16x16x32_bf16(
            false, aF.v, false, winF[nt][0].v, (short)0, biasF[nt], false,
            false);
    }
#pragma unroll
    for (int kc = 1; kc < NKIN; ++kc) {
      BF16x16 aF;
#pragma unroll
      for (int j = 0; j < 16; ++j) {
        const int K = kc * 32 + kbA + j + ((j >= 8) ? 8 : 0);
        aF.u[j] = s_inA[nl * H + K];
      }
#pragma unroll
      for (int nt = 0; nt < 4; ++nt)
        acc[nt] = __builtin_amdgcn_wmma_f32_16x16x32_bf16(
            false, aF.v, false, winF[nt][kc].v, (short)0, acc[nt], false, false);
    }
#pragma unroll
    for (int kc = 0; kc < 4; ++kc) {
      BF16x16 aF;
#pragma unroll
      for (int j = 0; j < 16; ++j) {
        const int K = kc * 32 + kbA + j + ((j >= 8) ? 8 : 0);
        aF.u[j] = s_memA[nl * H + K];
      }
#pragma unroll
      for (int nt = 0; nt < 4; ++nt)
        acc[nt] = __builtin_amdgcn_wmma_f32_16x16x32_bf16(
            false, aF.v, false, whhF[nt][kc].v, (short)0, acc[nt], false, false);
    }

    // -- per-wave gate nonlinearity, scatter to LDS --
#pragma unroll
    for (int nt = 0; nt < 4; ++nt) {
      const int col = wave * 64 + nt * 16 + nl;
#pragma unroll
      for (int j = 0; j < 8; ++j) {
        float v = acc[nt][j];
        v = (gsel == 2) ? fast_tanh(v) : sigm(v);
        const int row = j + (hi << 3);
        s_act[row * G + col] = v;
      }
    }
    __syncthreads();

    // -- elementwise LSTM state update (snntorch SLSTM, subtract reset) --
    float gi[8], gf[8], gg[8], go[8];
    {
      const float* base = s_act + srow * G + sh;
      const float4* pi = (const float4*)(base);
      const float4* pf = (const float4*)(base + 128);
      const float4* pg = (const float4*)(base + 256);
      const float4* po = (const float4*)(base + 384);
      float4 v0, v1;
      v0 = pi[0]; v1 = pi[1];
      gi[0]=v0.x; gi[1]=v0.y; gi[2]=v0.z; gi[3]=v0.w;
      gi[4]=v1.x; gi[5]=v1.y; gi[6]=v1.z; gi[7]=v1.w;
      v0 = pf[0]; v1 = pf[1];
      gf[0]=v0.x; gf[1]=v0.y; gf[2]=v0.z; gf[3]=v0.w;
      gf[4]=v1.x; gf[5]=v1.y; gf[6]=v1.z; gf[7]=v1.w;
      v0 = pg[0]; v1 = pg[1];
      gg[0]=v0.x; gg[1]=v0.y; gg[2]=v0.z; gg[3]=v0.w;
      gg[4]=v1.x; gg[5]=v1.y; gg[6]=v1.z; gg[7]=v1.w;
      v0 = po[0]; v1 = po[1];
      go[0]=v0.x; go[1]=v0.y; go[2]=v0.z; go[3]=v0.w;
      go[4]=v1.x; go[5]=v1.y; go[6]=v1.z; go[7]=v1.w;
    }
    unsigned short spkb[8];
#pragma unroll
    for (int j = 0; j < 8; ++j) {
      const float reset = (mem[j] > thr) ? thr : 0.f;   // prev-mem reset
      const float sn = gf[j] * syn[j] + gi[j] * gg[j];
      syn[j] = sn;
      const float mn = go[j] * fast_tanh(sn) - reset;
      mem[j] = mn;
      if constexpr (LAYER == 1) {
        const float spkv = (mn > thr) ? 1.f : 0.f;
        accum[j] += spkv;
        spkb[j] = f2bf(spkv);
      } else {
        accum[j] += mn;
      }
      s_memA[srow * H + sh + j] = f2bf(mn);
    }
    if constexpr (LAYER == 1) {
      uint4 pk;
      pk.x = (unsigned)spkb[0] | ((unsigned)spkb[1] << 16);
      pk.y = (unsigned)spkb[2] | ((unsigned)spkb[3] << 16);
      pk.z = (unsigned)spkb[4] | ((unsigned)spkb[5] << 16);
      pk.w = (unsigned)spkb[6] | ((unsigned)spkb[7] << 16);
      *(uint4*)(spk_out + ((size_t)t * B + r0 + srow) * H + sh) = pk;
    }
    __syncthreads();
  }

  if constexpr (LAYER == 1) {
#pragma unroll
    for (int j = 0; j < 8; ++j) atomicAdd(&spk_sum[sh + j], accum[j]);
  } else {
#pragma unroll
    for (int j = 0; j < 8; ++j)
      mem_out[(size_t)(r0 + srow) * H + sh + j] = accum[j] * (1.0f / 400.0f);
  }
}

// ---------------------------------------------------------------------------
// K6: final FC  out[b,nc] = mean_mem[b,:] . fc_w[nc,:] + fc_b[nc]
// ---------------------------------------------------------------------------
__global__ __launch_bounds__(256) void fc_kernel(
    const float* __restrict__ fm, const float* __restrict__ fw,
    const float* __restrict__ fb, float* __restrict__ out) {
  const int idx = blockIdx.x * blockDim.x + threadIdx.x;   // 8192
  const int b = idx >> 3, nc = idx & 7;
  float acc = fb[nc];
#pragma unroll 8
  for (int h = 0; h < 128; ++h) acc += fm[b * 128 + h] * fw[nc * 128 + h];
  out[idx] = acc;
}

// ---------------------------------------------------------------------------
extern "C" void kernel_launch(void* const* d_in, const int* in_sizes, int n_in,
                              void* d_out, int out_size, void* d_ws,
                              size_t ws_size, hipStream_t stream) {
  (void)in_sizes; (void)n_in; (void)out_size; (void)ws_size;
  const float* x     = (const float*)d_in[0];
  const float* Wih1  = (const float*)d_in[1];
  const float* Whh1  = (const float*)d_in[2];
  const float* bih1  = (const float*)d_in[3];
  const float* bhh1  = (const float*)d_in[4];
  const float* thr1  = (const float*)d_in[5];
  const float* Wih2  = (const float*)d_in[6];
  const float* Whh2  = (const float*)d_in[7];
  const float* bih2  = (const float*)d_in[8];
  const float* bhh2  = (const float*)d_in[9];
  const float* thr2  = (const float*)d_in[10];
  const float* bn1_g = (const float*)d_in[11];
  const float* bn1_b = (const float*)d_in[12];
  const float* bnh_g = (const float*)d_in[13];
  const float* bnh_b = (const float*)d_in[14];
  const float* fc_w  = (const float*)d_in[15];
  const float* fc_b  = (const float*)d_in[16];

  char* ws = (char*)d_ws;
  float*          stats     = (float*)(ws);                 //  32 f32
  float*          spk_sum   = (float*)(ws + 512);           // 128 f32
  float*          cb1       = (float*)(ws + 1024);          // 512 f32
  float*          cb2       = (float*)(ws + 3072);          // 512 f32
  unsigned short* W1s       = (unsigned short*)(ws + 8192);   //  32x512 bf16
  unsigned short* W2s       = (unsigned short*)(ws + 40960);  // 128x512 bf16
  float*          final_mem = (float*)(ws + 172032);          // 1024x128 f32
  unsigned short* spk1      = (unsigned short*)(ws + 696320); // 400x1024x128 bf16

  hipMemsetAsync(d_ws, 0, 1024, stream);   // stats + spike sums

  bn_stats_kernel<<<64, 256, 0, stream>>>(x, stats);
  prep1_kernel<<<1, 256, 0, stream>>>(stats, Wih1, bih1, bhh1, bn1_g, bn1_b,
                                      cb1, W1s);
  slstm_layer_kernel<1><<<64, 256, 0, stream>>>(
      x, (const unsigned short*)nullptr, Whh1, W1s, cb1, thr1, spk1, spk_sum,
      (float*)nullptr);
  prep2_kernel<<<1, 256, 0, stream>>>(spk_sum, Wih2, bih2, bhh2, bnh_g, bnh_b,
                                      cb2, W2s);
  slstm_layer_kernel<2><<<64, 256, 0, stream>>>(
      (const float*)nullptr, spk1, Whh2, W2s, cb2, thr2,
      (unsigned short*)nullptr, (float*)nullptr, final_mem);
  fc_kernel<<<32, 256, 0, stream>>>(final_mem, fc_w, fc_b, (float*)d_out);
}
